// MMSingleStreamBlock_15324443312829
// MI455X (gfx1250) — matrix operand order repaired and checked by reading, not compiled, not run on hardware
//
#include <hip/hip_runtime.h>

// ---------------------------------------------------------------------------
// MMSingleStreamBlock for MI455X (gfx1250, wave32, WMMA).
// Memory-bound workload (~1 GB HBM traffic/call @ 23.3 TB/s => ~40us floor);
// all GEMMs use v_wmma_f32_16x16x32_f16 (f32 accumulate), fp32 data streamed
// through LDS with global_load_async_to_lds_b128 and converted to f16 only in
// registers when building WMMA fragments. Stage addressing is incremental
// (no 64-bit multiplies in the inner loops).
// ---------------------------------------------------------------------------

typedef __attribute__((ext_vector_type(16))) _Float16 v16h;
typedef __attribute__((ext_vector_type(8)))  float    v8f;
typedef __attribute__((ext_vector_type(4)))  float    v4f;
typedef __attribute__((ext_vector_type(4)))  int      v4i;

#define HIDDEN    3072
#define HEADS     24
#define HEAD_DIM  128
#define MLPH      12288
#define NB1       (3*HIDDEN + MLPH)   // 21504
#define NBATCH    2
#define SEQ       1280
#define LIMG      1024
#define MROWS     (NBATCH*SEQ)        // 2560
#define K2TOT     (HIDDEN + MLPH)     // 15360

// ---------------- async global->LDS (guarded; falls back to vector copy) ---
#if defined(__has_builtin)
#  if __has_builtin(__builtin_amdgcn_global_load_async_to_lds_b128) && \
      __has_builtin(__builtin_amdgcn_s_wait_asynccnt)
#    define USE_ASYNC_LDS 1
#  endif
#endif
#ifndef USE_ASYNC_LDS
#  define USE_ASYNC_LDS 0
#endif

#if USE_ASYNC_LDS
typedef __attribute__((address_space(1))) v4i as1_v4i;   // global
typedef __attribute__((address_space(3))) v4i as3_v4i;   // LDS
#endif

__device__ __forceinline__ void copy16_g2s(float* lds_dst, const float* g_src) {
#if USE_ASYNC_LDS
  __builtin_amdgcn_global_load_async_to_lds_b128((as1_v4i*)g_src,
                                                 (as3_v4i*)lds_dst, 0, 0);
#else
  *(v4f*)lds_dst = *(const v4f*)g_src;
#endif
}

__device__ __forceinline__ void async_join() {
#if USE_ASYNC_LDS
  __builtin_amdgcn_s_wait_asynccnt(0);
#endif
}

// ---------------- WMMA helpers ---------------------------------------------
__device__ __forceinline__ v8f wmma_f16(v16h a, v16h b, v8f c) {
  return __builtin_amdgcn_wmma_f32_16x16x32_f16(false, a, false, b,
                                                (short)0, c, false, false);
}

__device__ __forceinline__ v8f vzero8() {
  v8f z;
#pragma unroll
  for (int i = 0; i < 8; ++i) z[i] = 0.0f;
  return z;
}

// A fragment (16x32 f16): lane m = lane&15 selects row; element e has
// k = hi*16 + e. rowk points at the lane's row, k-chunk base.
__device__ __forceinline__ v16h make_afrag(const float* rowk, int hi, float scale) {
  union { v16h v; _Float16 h[16]; } u;
  const float* p0 = rowk + hi * 16;
  const float* p1 = rowk + hi * 16 + 8;
#pragma unroll
  for (int i = 0; i < 8; ++i) {
    u.h[i]     = (_Float16)(p0[i] * scale);
    u.h[8 + i] = (_Float16)(p1[i] * scale);
  }
  return u.v;
}

// B fragment (32x16 f16), K-major source with row stride strideN:
// b[e] = src[(hi*16 + e)*strideN]; lane's column already folded into src.
__device__ __forceinline__ v16h make_bfrag_strided(const float* src, int strideN, int hi) {
  union { v16h v; _Float16 h[16]; } u;
  const float* p = src + (long)hi * 16 * strideN;
#pragma unroll
  for (int e = 0; e < 16; ++e) u.h[e] = (_Float16)p[e * strideN];
  return u.v;
}

// B fragment where the K run is contiguous in memory (e.g. K^T from K rows):
// b[e] = src[e] with src already offset by hi*16.
__device__ __forceinline__ v16h make_bfrag_contig(const float* src) {
  union { v16h v; _Float16 h[16]; } u;
#pragma unroll
  for (int e = 0; e < 16; ++e) u.h[e] = (_Float16)src[e];
  return u.v;
}

__device__ __forceinline__ float gelu_tanh(float x) {
  float x3 = x * x * x;
  float t  = tanhf(0.7978845608028654f * (x + 0.044715f * x3));
  return 0.5f * x * (1.0f + t);
}

// ---------------- kernel 0: mod = silu(vec) @ wmod + bmod ------------------
__global__ void __launch_bounds__(256)
mod_silu_kernel(const float* __restrict__ vec, const float* __restrict__ wmod,
                const float* __restrict__ bmod, float* __restrict__ mod) {
  __shared__ float sv[HIDDEN];
  int b = blockIdx.y;
  for (int i = threadIdx.x; i < HIDDEN; i += 256) {
    float v = vec[b * HIDDEN + i];
    sv[i] = v / (1.0f + __expf(-v));
  }
  __syncthreads();
  int col = blockIdx.x * 256 + threadIdx.x;   // 0..9215
  float acc = bmod[col];
  const float* wp = wmod + col;
  for (int k = 0; k < HIDDEN; ++k)
    acc = fmaf(sv[k], wp[(long)k * (3 * HIDDEN)], acc);
  mod[(long)b * (3 * HIDDEN) + col] = acc;
}

// ---------------- kernel 1: x_mod = LN(x)*(1+scale) + shift ----------------
__global__ void __launch_bounds__(256)
lnmod_kernel(const float* __restrict__ x, const float* __restrict__ mod,
             float* __restrict__ xmod) {
  __shared__ float r1[256], r2[256];
  int row = blockIdx.x;                 // 0..2559
  int b   = row / SEQ;
  const float* xr = x + (long)row * HIDDEN;
  float s1 = 0.0f, s2 = 0.0f;
  for (int i = threadIdx.x; i < HIDDEN; i += 256) {
    float v = xr[i];
    s1 += v; s2 += v * v;
  }
  r1[threadIdx.x] = s1; r2[threadIdx.x] = s2;
  __syncthreads();
  for (int off = 128; off > 0; off >>= 1) {
    if (threadIdx.x < off) { r1[threadIdx.x] += r1[threadIdx.x + off];
                             r2[threadIdx.x] += r2[threadIdx.x + off]; }
    __syncthreads();
  }
  float mu   = r1[0] * (1.0f / HIDDEN);
  float var  = r2[0] * (1.0f / HIDDEN) - mu * mu;
  float rinv = rsqrtf(var + 1e-6f);
  const float* shift = mod + (long)b * (3 * HIDDEN);
  const float* scale = shift + HIDDEN;
  for (int i = threadIdx.x; i < HIDDEN; i += 256) {
    float v = (xr[i] - mu) * rinv;
    xmod[(long)row * HIDDEN + i] = v * (1.0f + scale[i]) + shift[i];
  }
}

// ---------------- GEMM: C = act(A @ W + bias) [+ gated residual] -----------
// Block tile 64x128, K-stage 32, 8 waves (2x4), each wave a 32x32 subtile.
// A may be split at column splitK between two sources (for the concat GEMM).
__global__ void __launch_bounds__(256)
gemm_wmma_kernel(const float* __restrict__ A0, long sA0,
                 const float* __restrict__ A1, long sA1, int splitK,
                 int K, const float* __restrict__ W, int N,
                 const float* __restrict__ bias, float* __restrict__ C,
                 int geluStart,
                 const float* __restrict__ xres, const float* __restrict__ gate,
                 int rowsPerBatch) {
  __shared__ float As[2][64][32];
  __shared__ float Bs[2][32][128];

  int m0   = blockIdx.x * 64;     // M fastest => W panel reused from L2
  int n0   = blockIdx.y * 128;
  int tid  = threadIdx.x;
  int w    = tid >> 5;
  int lane = tid & 31;
  int hi   = lane >> 4;
  int ln16 = lane & 15;
  int wm   = (w >> 2) * 32;       // wave M offset inside tile
  int wn   = (w & 3) * 32;        // wave N offset inside tile

  // ---- per-thread fixed staging coordinates (no per-stage 64-bit muls) ----
  int aRow[2], aC4[2], bRow[4], bC4[4];
#pragma unroll
  for (int i = 0; i < 2; ++i) { int ch = tid + i * 256; aRow[i] = ch >> 3; aC4[i] = (ch & 7) * 4; }
#pragma unroll
  for (int i = 0; i < 4; ++i) { int ch = tid + i * 256; bRow[i] = ch >> 5; bC4[i] = (ch & 31) * 4; }

  const float* a0p[2]; const float* a1p[2]; const float* bp[4];
#pragma unroll
  for (int i = 0; i < 2; ++i) {
    a0p[i] = A0 + (long)(m0 + aRow[i]) * sA0 + aC4[i];
    a1p[i] = A1 ? (A1 + (long)(m0 + aRow[i]) * sA1 + aC4[i]) : a0p[i];
  }
#pragma unroll
  for (int i = 0; i < 4; ++i) bp[i] = W + (long)bRow[i] * N + n0 + bC4[i];

  auto loadStage = [&](int buf, int k0, long bOff) {
    bool inA0 = (k0 < splitK);
    long aOff = inA0 ? (long)k0 : (long)(k0 - splitK);
#pragma unroll
    for (int i = 0; i < 2; ++i)
      copy16_g2s(&As[buf][aRow[i]][aC4[i]], (inA0 ? a0p[i] : a1p[i]) + aOff);
#pragma unroll
    for (int i = 0; i < 4; ++i)
      copy16_g2s(&Bs[buf][bRow[i]][bC4[i]], bp[i] + bOff);
  };

  v8f acc[2][2];
#pragma unroll
  for (int i = 0; i < 2; ++i)
#pragma unroll
    for (int j = 0; j < 2; ++j) acc[i][j] = vzero8();

  const int NS = K / 32;
  long bOff = 0;

  loadStage(0, 0, 0);
  async_join();
  __syncthreads();

  for (int s = 0; s < NS; ++s) {
    int cur = s & 1;
    if (s + 1 < NS) {
      bOff += 32L * N;
      loadStage(cur ^ 1, (s + 1) * 32, bOff);
    }

    v16h a0 = make_afrag(&As[cur][wm + ln16][0],      hi, 1.0f);
    v16h a1 = make_afrag(&As[cur][wm + 16 + ln16][0], hi, 1.0f);
    v16h b0 = make_bfrag_strided(&Bs[cur][0][wn + ln16],      128, hi);
    v16h b1 = make_bfrag_strided(&Bs[cur][0][wn + 16 + ln16], 128, hi);
    acc[0][0] = wmma_f16(a0, b0, acc[0][0]);
    acc[0][1] = wmma_f16(a0, b1, acc[0][1]);
    acc[1][0] = wmma_f16(a1, b0, acc[1][0]);
    acc[1][1] = wmma_f16(a1, b1, acc[1][1]);

    async_join();
    __syncthreads();
  }

  // epilogue
#pragma unroll
  for (int i = 0; i < 2; ++i)
#pragma unroll
    for (int j = 0; j < 2; ++j) {
      int col = n0 + wn + j * 16 + ln16;
      float bc = bias[col];
#pragma unroll
      for (int r = 0; r < 8; ++r) {
        int row = m0 + wm + i * 16 + hi * 8 + r;
        float v = acc[i][j][r] + bc;
        if (geluStart >= 0 && col >= geluStart) v = gelu_tanh(v);
        if (gate) {
          int b = row / rowsPerBatch;
          v = xres[(long)row * N + col] + v * gate[(long)b * (3 * HIDDEN) + col];
        }
        C[(long)row * N + col] = v;
      }
    }
}

// ---------------- kernel 3: rms_norm(q/k) * gamma, RoPE on image rows ------
// One wave per (tensor, b, l, head) row of 128 dims; operates in-place on h.
__global__ void __launch_bounds__(256)
rmsrope_kernel(float* __restrict__ h, const float* __restrict__ qg,
               const float* __restrict__ kg, const float* __restrict__ fc,
               const float* __restrict__ fs) {
  int w = threadIdx.x >> 5, lane = threadIdx.x & 31;
  long idx = (long)blockIdx.x * 8 + w;       // 0 .. 2*2*1280*24-1
  int which = (int)(idx / (NBATCH * SEQ * HEADS));   // 0=q, 1=k
  int rem   = (int)(idx % (NBATCH * SEQ * HEADS));
  int b  = rem / (SEQ * HEADS); rem %= (SEQ * HEADS);
  int l  = rem / HEADS;
  int hd = rem % HEADS;

  float* p = h + (long)(b * SEQ + l) * NB1 + which * HIDDEN + hd * HEAD_DIM + lane * 4;
  v4f x = *(v4f*)p;
  float ss = x[0]*x[0] + x[1]*x[1] + x[2]*x[2] + x[3]*x[3];
#pragma unroll
  for (int off = 16; off >= 1; off >>= 1) ss += __shfl_xor(ss, off, 32);
  float rinv = rsqrtf(ss * (1.0f / HEAD_DIM) + 1e-6f);
  const float* g = (which == 0 ? qg : kg) + lane * 4;
  v4f y;
#pragma unroll
  for (int i = 0; i < 4; ++i) y[i] = x[i] * rinv * g[i];
  if (l < LIMG) {
    const float* cp = fc + (long)l * HEAD_DIM + lane * 4;
    const float* sp = fs + (long)l * HEAD_DIM + lane * 4;
    v4f c = *(const v4f*)cp, s = *(const v4f*)sp;
    v4f z;
    z[0] = y[0]*c[0] - y[1]*s[0];
    z[1] = y[1]*c[1] + y[0]*s[1];
    z[2] = y[2]*c[2] - y[3]*s[2];
    z[3] = y[3]*c[3] + y[2]*s[3];
    y = z;
  }
  *(v4f*)p = y;
}

// ---------------- kernel 4: flash attention --------------------------------
// Block = 128 q rows of one (b,head); 8 waves x 16 q rows. Key tiles of 32,
// K/V staged in LDS as f32, online softmax in registers, f16 WMMA for both
// Q@K^T (K-dim = head_dim) and P@V (K-dim = 32 keys).
__global__ void __launch_bounds__(256)
attn_flash_kernel(const float* __restrict__ h, float* __restrict__ attnOut) {
  __shared__ float Ks[32][HEAD_DIM];
  __shared__ float Vs[32][HEAD_DIM];
  __shared__ float Ps[8][16][32];

  int qt = blockIdx.x;               // 0..9
  int bh = blockIdx.y;               // 0..47
  int b  = bh / HEADS, hd = bh % HEADS;
  int tid  = threadIdx.x;
  int w    = tid >> 5;
  int lane = tid & 31;
  int hi   = lane >> 4;
  int ln16 = lane & 15;
  const float scale = 0.08838834764831845f;   // 1/sqrt(128)

  // Q fragments (4 K-chunks of 32 over head_dim), scale folded in.
  int qrow = qt * 128 + w * 16 + ln16;
  const float* qptr = h + (long)(b * SEQ + qrow) * NB1 + hd * HEAD_DIM;
  v16h qf[4];
#pragma unroll
  for (int c = 0; c < 4; ++c) qf[c] = make_afrag(qptr + c * 32, hi, scale);

  // per-thread staging pointers, advanced by 32 rows per key tile
  const float* kp[4]; const float* vp[4];
  float* ldsK[4]; float* ldsV[4];
#pragma unroll
  for (int i = 0; i < 4; ++i) {
    int ch = tid + i * 256;            // 0..1023
    int kr = ch >> 5, c4 = (ch & 31) * 4;
    const float* base = h + (long)(b * SEQ + kr) * NB1 + hd * HEAD_DIM + c4;
    kp[i] = base + HIDDEN;
    vp[i] = base + 2 * HIDDEN;
    ldsK[i] = &Ks[kr][c4];
    ldsV[i] = &Vs[kr][c4];
  }

  v8f O[8];
  float m_i[8], l_i[8];
#pragma unroll
  for (int s = 0; s < 8; ++s) O[s] = vzero8();
#pragma unroll
  for (int r = 0; r < 8; ++r) { m_i[r] = -3.0e38f; l_i[r] = 0.0f; }

  for (int kt = 0; kt < SEQ / 32; ++kt) {
    __syncthreads();                       // previous tile fully consumed
#pragma unroll
    for (int i = 0; i < 4; ++i) {          // stage 32 keys of K and V
      copy16_g2s(ldsK[i], kp[i]);
      copy16_g2s(ldsV[i], vp[i]);
      kp[i] += 32L * NB1;
      vp[i] += 32L * NB1;
    }
    async_join();
    __syncthreads();

    // S = Q @ K^T  (two 16x16 key subtiles)
    v8f s0 = vzero8(), s1 = vzero8();
#pragma unroll
    for (int c = 0; c < 4; ++c) {
      v16h kb0 = make_bfrag_contig(&Ks[ln16][c * 32 + hi * 16]);
      v16h kb1 = make_bfrag_contig(&Ks[16 + ln16][c * 32 + hi * 16]);
      s0 = wmma_f16(qf[c], kb0, s0);
      s1 = wmma_f16(qf[c], kb1, s1);
    }

    // online softmax per q-row (rows live as VGPR index r within 16-lane half)
#pragma unroll
    for (int r = 0; r < 8; ++r) {
      float v0 = s0[r], v1 = s1[r];
      float mx = fmaxf(v0, v1);
#pragma unroll
      for (int off = 8; off >= 1; off >>= 1) mx = fmaxf(mx, __shfl_xor(mx, off, 16));
      float mnew = fmaxf(m_i[r], mx);
      float al   = __expf(m_i[r] - mnew);
      float p0   = __expf(v0 - mnew);
      float p1   = __expf(v1 - mnew);
      float rs   = p0 + p1;
#pragma unroll
      for (int off = 8; off >= 1; off >>= 1) rs += __shfl_xor(rs, off, 16);
      l_i[r] = l_i[r] * al + rs;
      m_i[r] = mnew;
#pragma unroll
      for (int s = 0; s < 8; ++s) O[s][r] *= al;
      Ps[w][hi * 8 + r][ln16]      = p0;
      Ps[w][hi * 8 + r][16 + ln16] = p1;
    }

    // O += P @ V
    v16h pf = make_afrag(&Ps[w][ln16][0], hi, 1.0f);
#pragma unroll
    for (int s = 0; s < 8; ++s) {
      v16h vb = make_bfrag_strided(&Vs[0][s * 16 + ln16], HEAD_DIM, hi);
      O[s] = wmma_f16(pf, vb, O[s]);
    }
  }

  // write normalized output: attn[(b*SEQ+row), hd*128 + dim]
#pragma unroll
  for (int r = 0; r < 8; ++r) {
    float inv = 1.0f / l_i[r];
    int row = qt * 128 + w * 16 + hi * 8 + r;
    long base = (long)(b * SEQ + row) * HIDDEN + hd * HEAD_DIM;
#pragma unroll
    for (int s = 0; s < 8; ++s)
      attnOut[base + s * 16 + ln16] = O[s][r] * inv;
  }
}

// ---------------- launch ---------------------------------------------------
extern "C" void kernel_launch(void* const* d_in, const int* in_sizes, int n_in,
                              void* d_out, int out_size, void* d_ws, size_t ws_size,
                              hipStream_t stream) {
  const float* x    = (const float*)d_in[0];
  const float* vec  = (const float*)d_in[1];
  const float* w1   = (const float*)d_in[2];
  const float* b1   = (const float*)d_in[3];
  const float* w2   = (const float*)d_in[4];
  const float* b2   = (const float*)d_in[5];
  const float* wmod = (const float*)d_in[6];
  const float* bmod = (const float*)d_in[7];
  const float* qg   = (const float*)d_in[8];
  const float* kg   = (const float*)d_in[9];
  const float* fc   = (const float*)d_in[10];
  const float* fs   = (const float*)d_in[11];
  float* out = (float*)d_out;
  (void)in_sizes; (void)n_in; (void)out_size; (void)ws_size;

  float* ws   = (float*)d_ws;
  float* modw = ws;                     size_t o = (size_t)NBATCH * 3 * HIDDEN;
  float* xmod = ws + o;                 o += (size_t)MROWS * HIDDEN;
  float* hbuf = ws + o;                 o += (size_t)MROWS * NB1;
  float* attn = ws + o;                 /* o += (size_t)MROWS * HIDDEN; */

  // 0) modulation vector
  mod_silu_kernel<<<dim3((3 * HIDDEN) / 256, NBATCH), 256, 0, stream>>>(
      vec, wmod, bmod, modw);

  // 1) layer-norm + modulation
  lnmod_kernel<<<MROWS, 256, 0, stream>>>(x, modw, xmod);

  // 2) big fused QKV+MLP GEMM, gelu fused for MLP columns
  gemm_wmma_kernel<<<dim3(MROWS / 64, NB1 / 128), 256, 0, stream>>>(
      xmod, HIDDEN, nullptr, 0, HIDDEN,
      HIDDEN, w1, NB1, b1, hbuf,
      /*geluStart=*/3 * HIDDEN, nullptr, nullptr, 0);

  // 3) rms_norm(q/k) + RoPE, in place on h
  rmsrope_kernel<<<(2 * NBATCH * SEQ * HEADS) / 8, 256, 0, stream>>>(
      hbuf, qg, kg, fc, fs);

  // 4) flash attention
  attn_flash_kernel<<<dim3(SEQ / 128, NBATCH * HEADS), 256, 0, stream>>>(
      hbuf, attn);

  // 5) output GEMM over concat(attn, gelu(mlp)) + gated residual
  gemm_wmma_kernel<<<dim3(MROWS / 64, HIDDEN / 128), 256, 0, stream>>>(
      attn, HIDDEN, hbuf + 3 * HIDDEN, NB1, HIDDEN,
      K2TOT, w2, HIDDEN, b2, out,
      /*geluStart=*/-1, x, modw + 2 * HIDDEN, SEQ);
}
